// TestRmsnormGroupFp8QuantModel_39221641347418
// MI455X (gfx1250) — 8-bit matrix operand order repaired and checked by reading, compiled, not run on hardware
//
#include <hip/hip_runtime.h>
#include <hip/hip_bf16.h>
#include <stdint.h>


#define H_DIM   4096
#define BLK     128
#define KBLKS   (H_DIM / BLK)
#define FP8_MAX 448.0f
#define EPS     1e-6f

typedef int   v16i __attribute__((ext_vector_type(16)));
typedef float v8f  __attribute__((ext_vector_type(8)));

#if __has_builtin(__builtin_amdgcn_global_load_async_to_lds_b128)
#define USE_ASYNC 1
#else
#define USE_ASYNC 0
#endif

typedef int v4ig __attribute__((vector_size(16)));
typedef __attribute__((address_space(1))) v4ig gvec_t;
typedef __attribute__((address_space(3))) v4ig lvec_t;

__device__ __forceinline__ void wait_async0() {
#if __has_builtin(__builtin_amdgcn_s_wait_asynccnt)
    __builtin_amdgcn_s_wait_asynccnt(0);
#else
    asm volatile("s_wait_asynccnt 0" ::: "memory");
#endif
}

__device__ __forceinline__ void stage_async64(const unsigned char* gsrc,
                                              unsigned char* ldst) {
#if USE_ASYNC
#pragma unroll
    for (int i = 0; i < 4; ++i) {
        __builtin_amdgcn_global_load_async_to_lds_b128(
            (gvec_t*)(unsigned long long)(gsrc + 16 * i),
            (lvec_t*)(unsigned int)(unsigned long long)(ldst + 16 * i),
            0, 0);
    }
#else
    (void)gsrc; (void)ldst;
#endif
}

__device__ __forceinline__ unsigned int f32_to_e4m3(float x) {
    unsigned int u  = __float_as_uint(x);
    unsigned int sg = (u >> 24) & 0x80u;
    unsigned int au = u & 0x7FFFFFFFu;
    if (au >= 0x7F800000u) return sg | 0x7Fu;
    float a = __uint_as_float(au);
    if (a >= FP8_MAX) return sg | 0x7Eu;
    if (a < 0.015625f) {
        int m = (int)rintf(a * 512.0f);
        return sg | (unsigned int)m;
    }
    unsigned int lsb = (au >> 20) & 1u;
    au += 0x0007FFFFu + lsb;
    unsigned int expf = (au >> 23) - 127u + 7u;
    unsigned int mant = (au >> 20) & 7u;
    unsigned int r = (expf << 3) | mant;
    if (r > 0x7Eu) r = 0x7Eu;
    return sg | r;
}

__global__ __launch_bounds__(256)
void wq_transpose(const float* __restrict__ w, unsigned char* __restrict__ wqT) {
    __shared__ __align__(16) unsigned char tile[64 * 80];
    const int t  = threadIdx.x;
    const int k0 = blockIdx.y * 64;
    const int n0 = blockIdx.x * 64;
#pragma unroll
    for (int j = 0; j < 16; ++j) {
        int id = j * 256 + t;
        int kl = id >> 6, nl = id & 63;
        float v = w[(size_t)(k0 + kl) * H_DIM + (n0 + nl)];
        tile[nl * 80 + kl] = (unsigned char)f32_to_e4m3(v);
    }
    __syncthreads();
    int row = t >> 2, off = (t & 3) * 16;
    uint4 v = *(const uint4*)&tile[row * 80 + off];
    *(uint4*)(wqT + (size_t)(n0 + row) * H_DIM + k0 + off) = v;
}

__global__ __launch_bounds__(256)
void rms_quant(const float* __restrict__ src, const float* __restrict__ nw,
               float* __restrict__ resid_out, unsigned char* __restrict__ q,
               float* __restrict__ s_out, int apply_relu) {
    __shared__ float red[8];
    const int row  = blockIdx.x;
    const int t    = threadIdx.x;
    const int wave = t >> 5, lane = t & 31;
    const float* rp = src + (size_t)row * H_DIM + t * 16;

    float v[16];
#pragma unroll
    for (int c = 0; c < 4; ++c) {
        float4 f = ((const float4*)rp)[c];
        v[4*c+0] = f.x; v[4*c+1] = f.y; v[4*c+2] = f.z; v[4*c+3] = f.w;
    }
    if (apply_relu) {
#pragma unroll
        for (int i = 0; i < 16; ++i) v[i] = fmaxf(v[i], 0.0f);
    }
    if (resid_out) {
        float* op = resid_out + (size_t)row * H_DIM + t * 16;
#pragma unroll
        for (int c = 0; c < 4; ++c) {
            float4 f = make_float4(v[4*c+0], v[4*c+1], v[4*c+2], v[4*c+3]);
            ((float4*)op)[c] = f;
        }
    }
    float ss = 0.0f;
#pragma unroll
    for (int i = 0; i < 16; ++i) ss = fmaf(v[i], v[i], ss);
#pragma unroll
    for (int m = 16; m >= 1; m >>= 1) ss += __shfl_xor(ss, m, 32);
    if (lane == 0) red[wave] = ss;
    __syncthreads();
    float tot = red[0] + red[1] + red[2] + red[3] +
                red[4] + red[5] + red[6] + red[7];
    float rms = rsqrtf(tot * (1.0f / (float)H_DIM) + EPS);

    const float* wp = nw + t * 16;
    float y[16];
#pragma unroll
    for (int c = 0; c < 4; ++c) {
        float4 f = ((const float4*)wp)[c];
        y[4*c+0] = v[4*c+0] * rms * f.x;
        y[4*c+1] = v[4*c+1] * rms * f.y;
        y[4*c+2] = v[4*c+2] * rms * f.z;
        y[4*c+3] = v[4*c+3] * rms * f.w;
    }
    float am = 0.0f;
#pragma unroll
    for (int i = 0; i < 16; ++i) am = fmaxf(am, fabsf(y[i]));
    am = fmaxf(am, __shfl_xor(am, 1, 32));
    am = fmaxf(am, __shfl_xor(am, 2, 32));
    am = fmaxf(am, __shfl_xor(am, 4, 32));
    float s   = fmaxf(am, 1e-12f) * (1.0f / FP8_MAX);
    float inv = 1.0f / s;

    unsigned int b[4] = {0u, 0u, 0u, 0u};
#pragma unroll
    for (int i = 0; i < 16; ++i)
        b[i >> 2] |= f32_to_e4m3(y[i] * inv) << ((i & 3) * 8);
    uint4 pk; pk.x = b[0]; pk.y = b[1]; pk.z = b[2]; pk.w = b[3];
    ((uint4*)(q + (size_t)row * H_DIM))[t] = pk;
    if ((lane & 7) == 0) s_out[row * KBLKS + (t >> 3)] = s;
}

#if USE_ASYNC
#define LDS_TILES 2
#else
#define LDS_TILES 1
#endif

__global__ __launch_bounds__(256)
void gemm_fp8(const unsigned char* __restrict__ qa,
              const float* __restrict__ sa,
              const unsigned char* __restrict__ wqT,
              const float* __restrict__ ws,
              const float* __restrict__ rin,
              float* __restrict__ rout) {
    __shared__ __align__(16) uint4 ldsB[LDS_TILES * 128 * 8];

    const int tid  = threadIdx.x;
    const int wave = tid >> 5;
    const int lane = tid & 31;
    const int h    = lane >> 4;
    const int ln   = lane & 15;
    const int n0   = blockIdx.x * 128;
    const int m0   = blockIdx.y * 128 + wave * 16;
    const int nb   = blockIdx.x;

    v8f acc[8];
#pragma unroll
    for (int t = 0; t < 8; ++t)
#pragma unroll
        for (int r = 0; r < 8; ++r) acc[t][r] = 0.0f;

    const unsigned char* aRow = qa + (size_t)(m0 + ln) * H_DIM;
    const int cload = tid >> 1, hload = tid & 1;
    const unsigned char* bSrcBase =
        wqT + (size_t)(n0 + cload) * H_DIM + hload * 64;
    unsigned char* ldsByteBase =
        (unsigned char*)ldsB + cload * 128 + hload * 64;

#if USE_ASYNC
    stage_async64(bSrcBase, ldsByteBase);
#endif

    for (int kb = 0; kb < KBLKS; ++kb) {
#if USE_ASYNC
        wait_async0();
        __syncthreads();
        if (kb + 1 < KBLKS)
            stage_async64(bSrcBase + ((kb + 1) << 7),
                          ldsByteBase + ((kb + 1) & 1) * 16384);
        const uint4* ldsTile = ldsB + (kb & 1) * 1024;
#else
        {
            const uint4* src = (const uint4*)(bSrcBase + (kb << 7));
            uint4 x0 = src[0], x1 = src[1], x2 = src[2], x3 = src[3];
            uint4* dst = (uint4*)ldsByteBase;
            dst[0] = x0; dst[1] = x1; dst[2] = x2; dst[3] = x3;
        }
        if (kb + 1 < KBLKS)
            __builtin_prefetch(bSrcBase + ((kb + 1) << 7), 0, 0);
        __syncthreads();
        const uint4* ldsTile = ldsB;
#endif

        v16i afr;
        {
            const uint2* ap = (const uint2*)(aRow + (kb << 7));
#pragma unroll
            for (int p = 0; p < 8; ++p) {
                uint2 u = ap[(p >> 1) * 4 + h * 2 + (p & 1)];
                afr[2 * p + 0] = (int)u.x;
                afr[2 * p + 1] = (int)u.y;
            }
        }

        float wsk = ws[kb * KBLKS + nb];
        float sr[8];
#pragma unroll
        for (int r = 0; r < 8; ++r)
            sr[r] = sa[(size_t)(m0 + r + 8 * h) * KBLKS + kb] * wsk;

#pragma unroll
        for (int t = 0; t < 8; ++t) {
            const uint4* bp = &ldsTile[(t * 16 + ln) * 8 + h];
            uint4 b0 = bp[0], b1 = bp[2], b2 = bp[4], b3 = bp[6];
            v16i bfr;
            bfr[0]  = (int)b0.x; bfr[1]  = (int)b0.y;
            bfr[2]  = (int)b0.z; bfr[3]  = (int)b0.w;
            bfr[4]  = (int)b1.x; bfr[5]  = (int)b1.y;
            bfr[6]  = (int)b1.z; bfr[7]  = (int)b1.w;
            bfr[8]  = (int)b2.x; bfr[9]  = (int)b2.y;
            bfr[10] = (int)b2.z; bfr[11] = (int)b2.w;
            bfr[12] = (int)b3.x; bfr[13] = (int)b3.y;
            bfr[14] = (int)b3.z; bfr[15] = (int)b3.w;

            v8f zero;
#pragma unroll
            for (int r = 0; r < 8; ++r) zero[r] = 0.0f;
            v8f p8 = __builtin_amdgcn_wmma_f32_16x16x128_fp8_fp8(
                afr, bfr, (short)0, zero, false, false);
#pragma unroll
            for (int r = 0; r < 8; ++r)
                acc[t][r] = fmaf(sr[r], p8[r], acc[t][r]);
        }
#if !USE_ASYNC
        __syncthreads();
#endif
    }

#pragma unroll
    for (int t = 0; t < 8; ++t) {
        int col = n0 + t * 16 + ln;
#pragma unroll
        for (int r = 0; r < 8; ++r) {
            size_t idx = (size_t)(m0 + r + 8 * h) * H_DIM + col;
            rout[idx] = rin[idx] + acc[t][r];
        }
    }
}

__global__ __launch_bounds__(256)
void rms_final(const float* __restrict__ src, const float* __restrict__ nw,
               float* __restrict__ out) {
    __shared__ float red[8];
    const int row  = blockIdx.x;
    const int t    = threadIdx.x;
    const int wave = t >> 5, lane = t & 31;
    const float* rp = src + (size_t)row * H_DIM + t * 16;

    float v[16];
#pragma unroll
    for (int c = 0; c < 4; ++c) {
        float4 f = ((const float4*)rp)[c];
        v[4*c+0] = f.x; v[4*c+1] = f.y; v[4*c+2] = f.z; v[4*c+3] = f.w;
    }
    float ss = 0.0f;
#pragma unroll
    for (int i = 0; i < 16; ++i) ss = fmaf(v[i], v[i], ss);
#pragma unroll
    for (int m = 16; m >= 1; m >>= 1) ss += __shfl_xor(ss, m, 32);
    if (lane == 0) red[wave] = ss;
    __syncthreads();
    float tot = red[0] + red[1] + red[2] + red[3] +
                red[4] + red[5] + red[6] + red[7];
    float rms = rsqrtf(tot * (1.0f / (float)H_DIM) + EPS);

    const float* wp = nw + t * 16;
    float* op = out + (size_t)row * H_DIM + t * 16;
#pragma unroll
    for (int c = 0; c < 4; ++c) {
        float4 w = ((const float4*)wp)[c];
        float4 o;
        o.x = v[4*c+0] * rms * w.x;
        o.y = v[4*c+1] * rms * w.y;
        o.z = v[4*c+2] * rms * w.z;
        o.w = v[4*c+3] * rms * w.w;
        ((float4*)op)[c] = o;
    }
}

extern "C" void kernel_launch(void* const* d_in, const int* in_sizes, int n_in,
                              void* d_out, int out_size, void* d_ws, size_t ws_size,
                              hipStream_t stream) {
    (void)in_sizes; (void)n_in; (void)out_size; (void)ws_size;

    const float* x   = (const float*)d_in[0];
    const float* w[3]  = {(const float*)d_in[1], (const float*)d_in[3], (const float*)d_in[5]};
    const float* wsc[3] = {(const float*)d_in[2], (const float*)d_in[4], (const float*)d_in[6]};
    const float* nw[4] = {(const float*)d_in[7], (const float*)d_in[8],
                          (const float*)d_in[9], (const float*)d_in[10]};
    float* out = (float*)d_out;

    uint8_t* p = (uint8_t*)d_ws;
    size_t off = 0;
    auto carve = [&](size_t bytes) -> uint8_t* {
        uint8_t* r = p + off;
        off = (off + bytes + 255) & ~(size_t)255;
        return r;
    };
    const size_t HB = (size_t)H_DIM * H_DIM;
    unsigned char* wq[3];
    for (int i = 0; i < 3; ++i) wq[i] = carve(HB);
    unsigned char* qx = carve(HB);
    float* sx    = (float*)carve((size_t)H_DIM * KBLKS * 4);
    float* resid = (float*)carve(HB * 4);

    dim3 blk(256);
    dim3 gT(H_DIM / 64, H_DIM / 64);
    for (int i = 0; i < 3; ++i)
        wq_transpose<<<gT, blk, 0, stream>>>(w[i], wq[i]);

    dim3 gG(H_DIM / 128, H_DIM / 128);

    rms_quant<<<H_DIM, blk, 0, stream>>>(x, nw[0], resid, qx, sx, 1);
    gemm_fp8<<<gG, blk, 0, stream>>>(qx, sx, wq[0], wsc[0], resid, resid);
    rms_quant<<<H_DIM, blk, 0, stream>>>(resid, nw[1], nullptr, qx, sx, 0);
    gemm_fp8<<<gG, blk, 0, stream>>>(qx, sx, wq[1], wsc[1], resid, resid);
    rms_quant<<<H_DIM, blk, 0, stream>>>(resid, nw[2], nullptr, qx, sx, 0);
    gemm_fp8<<<gG, blk, 0, stream>>>(qx, sx, wq[2], wsc[2], resid, resid);
    rms_final<<<H_DIM, blk, 0, stream>>>(resid, nw[3], out);
}
